// PointNetSetAbstraction_38259568673339
// MI455X (gfx1250) — matrix operand order repaired and checked by reading, not compiled, not run on hardware
//
#include <hip/hip_runtime.h>
#include <hip/hip_bf16.h>

typedef _Float16 v16h __attribute__((ext_vector_type(16)));
typedef _Float16 h8   __attribute__((ext_vector_type(8)));
typedef float    v8f  __attribute__((ext_vector_type(8)));
typedef unsigned int u32x4 __attribute__((ext_vector_type(4)));
typedef int          i32x8 __attribute__((ext_vector_type(8)));
typedef int          i32x4 __attribute__((ext_vector_type(4)));

// Problem constants (from reference setup_inputs)
#define BATCH 16
#define NPTS  4096
#define SPTS  1024
#define KSAMP 32
#define DFEAT 64
#define RAD2  0.04f
#define MROWS (BATCH * SPTS * KSAMP)   // 524288

// ---------------------------------------------------------------------------
// 1) Farthest point sampling: one block per batch, 16 points per thread held
//    in VGPRs, LDS tree argmax per iteration. Emits fps_idx (int32) and
//    new_xyz in (B,3,S) transposed layout directly into d_out.
// ---------------------------------------------------------------------------
__global__ void fps_kernel(const float* __restrict__ xyz,
                           float* __restrict__ newXyz,
                           int* __restrict__ fpsIdx) {
  constexpr int T = 256, P = NPTS / T;   // 16 pts / thread
  const int b = blockIdx.x;
  const int t = threadIdx.x;
  __shared__ float rd[T];
  __shared__ int   ri[T];
  const float* xb = xyz + (size_t)b * 3 * NPTS;

  float px[P], py[P], pz[P], dist[P];
#pragma unroll
  for (int i = 0; i < P; ++i) {
    const int p = i * T + t;
    px[i] = xb[p]; py[i] = xb[NPTS + p]; pz[i] = xb[2 * NPTS + p];
    dist[i] = 1e10f;
  }

  int far = 0;
  for (int it = 0; it < SPTS; ++it) {
    // centroid coords: uniform (scalar) loads, broadcast to all lanes
    const float cx = xb[far], cy = xb[NPTS + far], cz = xb[2 * NPTS + far];
    if (t == 0) {
      fpsIdx[b * SPTS + it] = far;
      newXyz[(b * 3 + 0) * SPTS + it] = cx;
      newXyz[(b * 3 + 1) * SPTS + it] = cy;
      newXyz[(b * 3 + 2) * SPTS + it] = cz;
    }
    float bd = -1.0f; int bi = 0;
#pragma unroll
    for (int i = 0; i < P; ++i) {
      const float dx = px[i] - cx, dy = py[i] - cy, dz = pz[i] - cz;
      const float d = dx * dx + dy * dy + dz * dz;
      dist[i] = fminf(dist[i], d);
      const int p = i * T + t;           // ascending, so strict > keeps lowest idx
      if (dist[i] > bd) { bd = dist[i]; bi = p; }
    }
    rd[t] = bd; ri[t] = bi;
    __syncthreads();
    for (int off = T / 2; off > 0; off >>= 1) {
      if (t < off) {
        const float od = rd[t + off]; const int oi = ri[t + off];
        if (od > rd[t] || (od == rd[t] && oi < ri[t])) { rd[t] = od; ri[t] = oi; }
      }
      __syncthreads();
    }
    far = ri[0];
    __syncthreads();   // everyone has read ri[0] before next iteration clobbers
  }
}

// ---------------------------------------------------------------------------
// 2) Ball query: one thread per query; 32-entry nearest list kept in LDS
//    (replace-current-worst). Order within K is irrelevant downstream (BN
//    stats + max-pool are permutation invariant), only the selected multiset
//    matters. Out-of-ball entries replaced by the nearest index (== idx[...,0]
//    semantics of the reference; the nearest is the centroid itself, dist 0).
// ---------------------------------------------------------------------------
__global__ void ball_kernel(const float* __restrict__ xyz,
                            const float* __restrict__ newXyz,
                            int* __restrict__ idxBuf) {
  constexpr int T = 128, STRIDE = KSAMP + 1;   // pad to dodge bank conflicts
  __shared__ float cd[T * STRIDE];
  __shared__ int   ci[T * STRIDE];
  const int t = threadIdx.x;
  const int b = blockIdx.x >> 3;
  const int s = ((blockIdx.x & 7) << 7) + t;
  float* myd = &cd[t * STRIDE];
  int*   myi = &ci[t * STRIDE];
  const float* xb = xyz + (size_t)b * 3 * NPTS;
  const float qx = newXyz[(b * 3 + 0) * SPTS + s];
  const float qy = newXyz[(b * 3 + 1) * SPTS + s];
  const float qz = newXyz[(b * 3 + 2) * SPTS + s];

  float worst = -1.0f; int wpos = 0;
  for (int j = 0; j < KSAMP; ++j) {
    const float dx = xb[j] - qx, dy = xb[NPTS + j] - qy, dz = xb[2 * NPTS + j] - qz;
    const float d = dx * dx + dy * dy + dz * dz;
    myd[j] = d; myi[j] = j;
    if (d > worst) { worst = d; wpos = j; }
  }
  for (int j = KSAMP; j < NPTS; ++j) {
    const float dx = xb[j] - qx, dy = xb[NPTS + j] - qy, dz = xb[2 * NPTS + j] - qz;
    const float d = dx * dx + dy * dy + dz * dz;
    if (d < worst) {
      myd[wpos] = d; myi[wpos] = j;
      worst = -1.0f;
#pragma unroll
      for (int e = 0; e < KSAMP; ++e)
        if (myd[e] > worst) { worst = myd[e]; wpos = e; }
    }
  }
  // nearest entry = fill index for out-of-ball slots
  float bestd = myd[0]; int besti = myi[0];
  for (int e = 1; e < KSAMP; ++e)
    if (myd[e] < bestd) { bestd = myd[e]; besti = myi[e]; }
  int* outp = idxBuf + ((size_t)(b * SPTS) + s) * KSAMP;
  for (int e = 0; e < KSAMP; ++e)
    outp[e] = (myd[e] <= RAD2) ? myi[e] : besti;
}

// ---------------------------------------------------------------------------
// 3) WMMA GEMM layer. MODE 0: fused gather+concat (xyz_norm | points) ->
//    f16 LDS tiles, Cin=67 zero-padded to 96. MODE 1: the previous layer's
//    raw pre-BN tile (128 rows x 64 ch f16, fully contiguous) is DMA'd into
//    LDS by the Tensor Data Mover (overlapping the VALU weight conversion),
//    then BN scale/shift + ReLU are fused into the A-tile staging.
//    Per-channel sum/sumsq accumulated (LDS atomics -> 1 global atomic per
//    channel per block) for the batch-norm statistics of THIS layer's output.
// ---------------------------------------------------------------------------
template <int MODE, int CIN, int CPAD, int NOUT>
__global__ void mlp_gemm_kernel(const float* __restrict__ xyz,
                                const float* __restrict__ points,
                                const float* __restrict__ newXyz,
                                const int* __restrict__ idxBuf,
                                const _Float16* __restrict__ hin,
                                const float* __restrict__ bnScale,
                                const float* __restrict__ bnShift,
                                const float* __restrict__ w,
                                const float* __restrict__ bias,
                                _Float16* __restrict__ hout,
                                float* __restrict__ gSum,
                                float* __restrict__ gSq) {
  constexpr int ROWS = 128, T = 256;
  __shared__ __align__(16) _Float16 aLds[ROWS * CPAD];
  __shared__ __align__(16) _Float16 wLds[NOUT * CPAD];
  __shared__ __align__(16) _Float16 hStage[(MODE == 1) ? ROWS * CIN : 1];
  __shared__ float bLds[NOUT];
  __shared__ float sSum[NOUT], sSq[NOUT];

  const int t = threadIdx.x;
  const int row0 = blockIdx.x * ROWS;

  if (t == 0) __builtin_prefetch(w, 0, 0);   // global_prefetch: warm weight rows

  // ---- MODE 1: TDM async copy of the contiguous activation tile into LDS ----
  if (MODE == 1) {
    if (t < 32) {   // one wave issues the descriptor (EXEC all-ones in wave 0)
      const unsigned long long gaddr =
          (unsigned long long)(const void*)(hin + (size_t)row0 * CIN);
      const unsigned ldsOff = (unsigned)(size_t)hStage;   // LDS byte offset
      const unsigned n = ROWS * CIN;                      // 8192 f16 elements
      // D# group 0: count=1 | lds_addr | global_addr[56:0] | type=2 (image)
      u32x4 g0 = { 1u, ldsOff, (unsigned)gaddr,
                   ((unsigned)(gaddr >> 32) & 0x01FFFFFFu) | (2u << 30) };
      // D# group 1: 1-D tile, data_size=2B (code 1), no multicast/pad/iterate
      i32x8 g1 = { (int)(1u << 16),          // data_size
                   (int)((n & 0xFFFFu) << 16),        // tensor_dim0[15:0]
                   (int)((n >> 16) | (1u << 16)),     // tensor_dim0[31:16] | tensor_dim1[15:0]
                   (int)((n & 0xFFFFu) << 16),        // tile_dim0
                   1,                                  // tile_dim1=1, tile_dim2=0
                   (int)n,                             // tensor_dim0_stride lo
                   0, 0 };
      i32x4 gz4 = { 0, 0, 0, 0 };
      i32x8 gz8 = { 0, 0, 0, 0, 0, 0, 0, 0 };
      // clang-23 / amdgpu-toolchain 6-arg form:
      // (g0 u32x4, g1 i32x8, g2 i32x4, g3 i32x4, g4 i32x8, cpol)
      __builtin_amdgcn_tensor_load_to_lds(g0, g1, gz4, gz4, gz8, 0);
      __builtin_amdgcn_s_wait_tensorcnt(0);
    }
  }

  // ---- stage activations (gather / BN+ReLU fused) ----
  if (MODE == 0) {
    for (int e = t; e < ROWS * CPAD; e += T) {
      const int r = e / CPAD, c = e - r * CPAD;
      const int m = row0 + r;
      float v;
      const int b = m >> 15;             // S*K = 32768 rows per batch
      const int rem = m & 32767;
      const int s = rem >> 5;
      const int pi = idxBuf[m];
      if (c < 3)
        v = xyz[(b * 3 + c) * NPTS + pi] - newXyz[(b * 3 + c) * SPTS + s];
      else if (c < CIN)
        v = points[(b * DFEAT + (c - 3)) * NPTS + pi];
      else
        v = 0.0f;
      aLds[e] = (_Float16)v;
    }
  }
  // ---- stage weights (f32 -> f16, zero pad Cin -> CPAD) ----
  for (int e = t; e < NOUT * CPAD; e += T) {
    const int o = e / CPAD, c = e - o * CPAD;
    wLds[e] = (c < CIN) ? (_Float16)w[o * CIN + c] : (_Float16)0.0f;
  }
  if (t < NOUT) { bLds[t] = bias[t]; sSum[t] = 0.0f; sSq[t] = 0.0f; }
  if (MODE == 1) {
    __syncthreads();   // hStage tile (TDM) visible to all waves
    for (int e = t; e < ROWS * CIN; e += T) {   // CPAD == CIN here
      const int c = e & (CIN - 1);
      const float x = (float)hStage[e];
      aLds[e] = (_Float16)fmaxf(bnScale[c] * x + bnShift[c], 0.0f);
    }
  }
  __syncthreads();

  // ---- WMMA compute: 8 waves, one 16-row M-tile each ----
  const int wave = t >> 5, lane = t & 31, laneLo = lane & 15;
  const bool hi = lane >= 16;
  const int rowA = wave * 16 + laneLo;
#pragma unroll
  for (int nt = 0; nt < NOUT / 16; ++nt) {
    v8f acc = {};
    const int col = nt * 16 + laneLo;
#pragma unroll
    for (int kc = 0; kc < CPAD / 32; ++kc) {
      // A 16x32 f16 fragment (ISA 7.12.2): lo lanes K=+0..7,+16..23; hi K=+8..15,+24..31
      const int ka = kc * 32 + (hi ? 8 : 0);
      const h8 alo = *(const h8*)&aLds[rowA * CPAD + ka];
      const h8 ahi = *(const h8*)&aLds[rowA * CPAD + ka + 16];
      // B 32x16: col per lane, lo lanes K=+0..7,+16..23; hi K=+8..15,+24..31
      const int kb = kc * 32 + (hi ? 8 : 0);
      const h8 blo = *(const h8*)&wLds[col * CPAD + kb];
      const h8 bhi = *(const h8*)&wLds[col * CPAD + kb + 16];
      v16h A, Bm;
#pragma unroll
      for (int i = 0; i < 8; ++i) {
        A[i] = alo[i];  A[i + 8] = ahi[i];
        Bm[i] = blo[i]; Bm[i + 8] = bhi[i];
      }
      acc = __builtin_amdgcn_wmma_f32_16x16x32_f16(
          /*neg_a=*/false, A, /*neg_b=*/false, Bm,
          /*c_mod=*/(short)0, acc, /*reuse_a=*/false, /*reuse_b=*/false);
    }
    // bias, store f16 (base pointer + constant offsets -> folded into the
    // 24-bit instruction offset, no per-element 64-bit address math), stats
    const float bb = bLds[col];
    float s1 = 0.0f, s2 = 0.0f;
    const int mbase = row0 + wave * 16 + (hi ? 8 : 0);
    _Float16* outp = hout + (size_t)mbase * NOUT + col;
#pragma unroll
    for (int r = 0; r < 8; ++r) {
      const float v = acc[r] + bb;
      outp[r * NOUT] = (_Float16)v;
      s1 += v; s2 += v * v;
    }
    atomicAdd(&sSum[col], s1);
    atomicAdd(&sSq[col], s2);
  }
  __syncthreads();
  if (t < NOUT) {
    atomicAdd(&gSum[t], sSum[t]);
    atomicAdd(&gSq[t], sSq[t]);
  }
}

// ---------------------------------------------------------------------------
// 4) BN statistics -> per-channel scale/shift
// ---------------------------------------------------------------------------
__global__ void bn_stats_kernel(const float* __restrict__ gSum,
                                const float* __restrict__ gSq,
                                const float* __restrict__ gamma,
                                const float* __restrict__ beta,
                                float* __restrict__ scale,
                                float* __restrict__ shift,
                                int n, float invCnt) {
  const int t = threadIdx.x;
  if (t < n) {
    const float mean = gSum[t] * invCnt;
    const float var = gSq[t] * invCnt - mean * mean;
    const float sc = gamma[t] * rsqrtf(var + 1e-5f);
    scale[t] = sc;
    shift[t] = beta[t] - mean * sc;
  }
}

__global__ void zero_kernel(float* __restrict__ p, int n) {
  const int i = blockIdx.x * blockDim.x + threadIdx.x;
  if (i < n) p[i] = 0.0f;
}

// ---------------------------------------------------------------------------
// 5) Final BN+ReLU + max over K=32 samples, store (B,128,S) transposed
// ---------------------------------------------------------------------------
__global__ void finalize_kernel(const _Float16* __restrict__ h3,
                                const float* __restrict__ scale,
                                const float* __restrict__ shift,
                                float* __restrict__ outPts) {
  constexpr int C = 128;
  const int c = threadIdx.x;
  const int s = blockIdx.x;
  const int b = blockIdx.y;
  const size_t m0 = ((size_t)(b * SPTS) + s) * KSAMP;
  const float sc = scale[c], sh = shift[c];
  const _Float16* hp = h3 + m0 * C + c;   // constant k*C offsets fold into insts
  float mx = 0.0f;                        // ReLU outputs are >= 0
#pragma unroll
  for (int k = 0; k < KSAMP; ++k) {
    float v = sc * (float)hp[k * C] + sh;
    v = fmaxf(v, 0.0f);
    mx = fmaxf(mx, v);
  }
  outPts[((size_t)b * C + c) * SPTS + s] = mx;
}

// ---------------------------------------------------------------------------
extern "C" void kernel_launch(void* const* d_in, const int* in_sizes, int n_in,
                              void* d_out, int out_size, void* d_ws, size_t ws_size,
                              hipStream_t stream) {
  const float* xyz    = (const float*)d_in[0];
  const float* points = (const float*)d_in[1];
  const float* w0 = (const float*)d_in[2];
  const float* b0 = (const float*)d_in[3];
  const float* g0 = (const float*)d_in[4];
  const float* be0 = (const float*)d_in[5];
  const float* w1 = (const float*)d_in[6];
  const float* b1 = (const float*)d_in[7];
  const float* g1 = (const float*)d_in[8];
  const float* be1 = (const float*)d_in[9];
  const float* w2 = (const float*)d_in[10];
  const float* b2 = (const float*)d_in[11];
  const float* g2 = (const float*)d_in[12];
  const float* be2 = (const float*)d_in[13];

  // outputs: new_xyz (16*3*1024 f32) | new_points (16*128*1024 f32) | fps_idx (16*1024 i32)
  float* out = (float*)d_out;
  float* newXyz = out;
  float* newPts = out + BATCH * 3 * SPTS;
  int*   fpsIdx = (int*)(out + BATCH * 3 * SPTS + (size_t)BATCH * 128 * SPTS);

  // workspace: idx (2 MiB) | h1 (64 MiB f16, 524288x64) | h3 (128 MiB f16, 524288x128) | stats
  char* ws = (char*)d_ws;
  int* idxBuf = (int*)ws;
  _Float16* h1 = (_Float16*)(ws + (size_t)(2u << 20));
  _Float16* h3 = (_Float16*)(ws + (size_t)(2u << 20) + ((size_t)64u << 20));
  float* stats = (float*)(ws + (size_t)(2u << 20) + ((size_t)64u << 20) + ((size_t)128u << 20));
  float* st0 = stats;          // layout per layer: sum[128] | sumsq[128] | scale[128] | shift[128]
  float* st1 = stats + 512;
  float* st2 = stats + 1024;

  const float invCnt = 1.0f / (float)MROWS;

  zero_kernel<<<6, 256, 0, stream>>>(stats, 1536);
  fps_kernel<<<BATCH, 256, 0, stream>>>(xyz, newXyz, fpsIdx);
  ball_kernel<<<128, 128, 0, stream>>>(xyz, newXyz, idxBuf);

  // layer 0: gather+concat (67 -> pad 96) -> 64, WMMA f16/f32
  mlp_gemm_kernel<0, 67, 96, 64><<<MROWS / 128, 256, 0, stream>>>(
      xyz, points, newXyz, idxBuf, nullptr, nullptr, nullptr,
      w0, b0, h1, st0 + 0, st0 + 128);
  bn_stats_kernel<<<1, 128, 0, stream>>>(st0, st0 + 128, g0, be0,
                                         st0 + 256, st0 + 384, 64, invCnt);
  // layer 1: 64 -> 64 (in-place safe: each block reads/writes only its rows)
  mlp_gemm_kernel<1, 64, 64, 64><<<MROWS / 128, 256, 0, stream>>>(
      nullptr, nullptr, nullptr, nullptr, h1, st0 + 256, st0 + 384,
      w1, b1, h1, st1 + 0, st1 + 128);
  bn_stats_kernel<<<1, 128, 0, stream>>>(st1, st1 + 128, g1, be1,
                                         st1 + 256, st1 + 384, 64, invCnt);
  // layer 2: 64 -> 128
  mlp_gemm_kernel<1, 64, 64, 128><<<MROWS / 128, 256, 0, stream>>>(
      nullptr, nullptr, nullptr, nullptr, h1, st1 + 256, st1 + 384,
      w2, b2, h3, st2 + 0, st2 + 128);
  bn_stats_kernel<<<1, 128, 0, stream>>>(st2, st2 + 128, g2, be2,
                                         st2 + 256, st2 + 384, 128, invCnt);

  finalize_kernel<<<dim3(SPTS, BATCH), 128, 0, stream>>>(h3, st2 + 256, st2 + 384, newPts);
}
